// SpectralStructureAlignment1d_19198503813367
// MI455X (gfx1250) — matrix operand order repaired and checked by reading, not compiled, not run on hardware
//
#include <hip/hip_runtime.h>
#include <hip/hip_bf16.h>
#include <math.h>

// ---------------------------------------------------------------------------
// SpectralStructureAlignment1d for MI455X (gfx1250, wave32, WMMA)
//   N=32, C=64, L=8192, F=256, HOP=128, NUM=63 frames, FR=129 bins, LR=4097
// ---------------------------------------------------------------------------

#define N_SIG   2048        // N*C
#define LSIG    8192
#define FLEN    256
#define HOP     128
#define NFRM    63
#define FR      129
#define NCOL    258         // [0,129): Re cols, [129,258): Im cols
#define NTILES  17          // ceil(258/16) -> padded to 272 cols
#define EPSV    1e-8f
#define PI_F    3.14159265358979323846f
#define TWO_PI  6.28318530717958647692f

typedef _Float16 v16h __attribute__((ext_vector_type(16)));
typedef float    v8f  __attribute__((ext_vector_type(8)));

// ------------------------------- helpers -----------------------------------
__device__ __forceinline__ float2 cmul(float2 a, float2 b) {
    return make_float2(a.x * b.x - a.y * b.y, a.x * b.y + a.y * b.x);
}

// ------------------------- 1) per-signal means -----------------------------
__global__ __launch_bounds__(256) void mean_kernel(const float* __restrict__ x,
                                                   float* __restrict__ means) {
    __shared__ float red[256];
    const int sig = blockIdx.x;
    const float* p = x + (size_t)sig * LSIG;
    float s = 0.f;
    for (int i = threadIdx.x; i < LSIG; i += 256) s += p[i];
    red[threadIdx.x] = s;
    __syncthreads();
    for (int o = 128; o > 0; o >>= 1) {
        if (threadIdx.x < o) red[threadIdx.x] += red[threadIdx.x + o];
        __syncthreads();
    }
    if (threadIdx.x == 0) means[sig] = red[0] * (1.0f / (float)LSIG);
}

// --------- 2) window-folded DFT matrix, pre-swizzled to B fragments --------
// B fragment layout this code uses (32x16 B, wave32; contiguous k, same order as A):
//   lanes 0-15 : N = lane,     elements j=0..15 hold K = 0..15
//   lanes 16-31: N = lane-16,  elements j=0..15 hold K = 16..31
// Stored hi/lo _Float16 pairs: idx = ((ntile*8 + kstep)*32 + lane)*16 + j
__global__ __launch_bounds__(256) void dft_init_kernel(_Float16* __restrict__ Bhi,
                                                       _Float16* __restrict__ Blo) {
    const int tid = blockIdx.x * 256 + threadIdx.x;
    const int total = NTILES * 8 * 32 * 16;
    if (tid >= total) return;
    const int j    = tid & 15;
    const int lane = (tid >> 4) & 31;
    const int ks   = (tid >> 9) & 7;
    const int nt   = tid >> 12;

    const int t   = ks * 32 + ((lane >> 4) << 4) + j;   // time index 0..255
    const int col = nt * 16 + (lane & 15);              // output column

    // normalized periodic Hann: ||w||^2 = 256*3/8 = 96
    const float wn = 0.5f * (1.0f - cosf(TWO_PI * (float)t * (1.0f / 256.0f)))
                     * 0.10206207261596577f;            // 1/sqrt(96)
    float v = 0.0f;
    if (col < FR) {                 // real part: cos
        const int ph = (col * t) & 255;                 // exact phase reduction
        v = wn * cosf(TWO_PI * (float)ph * (1.0f / 256.0f));
    } else if (col < NCOL) {        // imag part: -sin
        const int ph = ((col - FR) * t) & 255;
        v = -wn * sinf(TWO_PI * (float)ph * (1.0f / 256.0f));
    }
    const _Float16 hi = (_Float16)v;
    Bhi[tid] = hi;
    Blo[tid] = (_Float16)(v - (float)hi);
}

// ------- 3) Welch periodogram as split-f16 WMMA GEMM (one wave/signal) -----
// A fragment layout this code uses (16x32 16-bit A, wave32; contiguous k, same order as B):
//   lanes 0-15 : M = lane,    element j -> K = j
//   lanes 16-31: M = lane-16, element j -> K = j + 16
// C/D layout (v8f): lane holds col N = lane&15, rows r + 8*(lane>>4)
//
// The mtile/chunk/ks loops are kept rolled (#pragma unroll 1): round-2 disasm
// showed LLVM fully unrolling them, hoisting ~160 B-fragment loads and
// spilling them to scratch inside the WMMA loop. Rolled, the live set is
// ah/al (16 VGPRs) + bh/bl x4 (64) + acc (32): no spills.
__global__ __launch_bounds__(32) void welch_wmma_kernel(const float* __restrict__ x,
                                                        const float* __restrict__ means,
                                                        const _Float16* __restrict__ Bhi,
                                                        const _Float16* __restrict__ Blo,
                                                        float* __restrict__ P_hat) {
    __shared__ float sig[LSIG];                       // 32 KB centered signal
    __shared__ __align__(32) _Float16 Afrag[2][8][32][16];  // 16 KB, per-mtile
    __shared__ float SQ[NTILES * 16];

    const int s0   = blockIdx.x;
    const int lane = threadIdx.x;
    const float* xp = x + (size_t)s0 * LSIG;
    const float mu  = means[s0];

    for (int i = lane; i < LSIG; i += 32) sig[i] = xp[i] - mu;
    for (int i = lane; i < NTILES * 16; i += 32) SQ[i] = 0.0f;
    __syncthreads();

    const int row   = lane & 15;
    const int khalf = lane >> 4;

#pragma unroll 1
    for (int mtile = 0; mtile < 4; ++mtile) {
        const int  frame = mtile * 16 + row;
        const bool valid = frame < NFRM;
        const int  fbase = frame * HOP;

        // stage this mtile's 8 k-step A fragments (hi/lo split) into LDS
        __syncthreads();
#pragma unroll 1
        for (int ks = 0; ks < 8; ++ks) {
            v16h ah, al;
#pragma unroll
            for (int j = 0; j < 16; ++j) {
                const int tl = j + 16 * khalf;
                const float v = valid ? sig[fbase + ks * 32 + tl] : 0.0f;
                const _Float16 hi = (_Float16)v;
                ah[j] = hi;
                al[j] = (_Float16)(v - (float)hi);
            }
            *(v16h*)Afrag[0][ks][lane] = ah;
            *(v16h*)Afrag[1][ks][lane] = al;
        }
        __syncthreads();

        // output-column tiles in chunks of 4 (accumulators stay in VGPRs)
#pragma unroll 1
        for (int chunk = 0; chunk < 5; ++chunk) {
            v8f acc[4] = {};
#pragma unroll 1
            for (int ks = 0; ks < 8; ++ks) {
                const v16h ah = *(const v16h*)Afrag[0][ks][lane];
                const v16h al = *(const v16h*)Afrag[1][ks][lane];
#pragma unroll
                for (int u = 0; u < 4; ++u) {
                    const int nt = chunk * 4 + u;     // uniform across wave
                    if (nt < NTILES) {
                        const size_t bidx =
                            ((((size_t)nt * 8 + ks) * 32 + lane) << 4);
                        const v16h bh = *(const v16h*)(Bhi + bidx);
                        const v16h bl = *(const v16h*)(Blo + bidx);
                        // fp32-accurate product via 3-term f16 split
                        acc[u] = __builtin_amdgcn_wmma_f32_16x16x32_f16(
                            false, ah, false, bh, (short)0, acc[u], false, false);
                        acc[u] = __builtin_amdgcn_wmma_f32_16x16x32_f16(
                            false, ah, false, bl, (short)0, acc[u], false, false);
                        acc[u] = __builtin_amdgcn_wmma_f32_16x16x32_f16(
                            false, al, false, bh, (short)0, acc[u], false, false);
                    }
                }
            }
#pragma unroll
            for (int u = 0; u < 4; ++u) {
                const int nt = chunk * 4 + u;
                if (nt < NTILES) {
                    float ss = 0.0f;
#pragma unroll
                    for (int r = 0; r < 8; ++r) ss += acc[u][r] * acc[u][r];
                    atomicAdd(&SQ[nt * 16 + row], ss);   // ds_add_f32
                }
            }
        }
    }
    __syncthreads();

    float* Pp = P_hat + (size_t)s0 * FR;
    for (int k = lane; k < FR; k += 32)
        Pp[k] = (SQ[k] + SQ[FR + k]) * (1.0f / (float)NFRM);  // Re^2+Im^2 mean
}

// ------------------ 4) consensus spectrum over the batch -------------------
__global__ __launch_bounds__(256) void pref_kernel(const float* __restrict__ P_hat,
                                                   float* __restrict__ P_ref) {
    const int idx = blockIdx.x * 256 + threadIdx.x;   // c*129 + k
    if (idx >= 64 * FR) return;
    float s = 0.f;
    for (int n = 0; n < 32; ++n)
        s += sqrtf(fmaxf(P_hat[(size_t)n * 64 * FR + idx], EPSV));
    const float m = s * (1.0f / 32.0f);
    P_ref[idx] = m * m;
}

// --------------------------- 5) Wiener gain --------------------------------
__global__ __launch_bounds__(256) void hwelch_kernel(const float* __restrict__ P_hat,
                                                     const float* __restrict__ P_ref,
                                                     float* __restrict__ Hw) {
    const int idx = blockIdx.x * 256 + threadIdx.x;   // (n*64+c)*129 + k
    if (idx >= N_SIG * FR) return;
    const int ck = idx % (64 * FR);
    Hw[idx] = sqrtf(fmaxf(P_ref[ck], EPSV) / fmaxf(P_hat[idx], EPSV));
}

// --------- 6) full-length rfft -> filter -> irfft (LDS-resident) -----------
// twd[m] = e^{-i*pi*m/4096}; stage-s twiddle is twd[j << (12-s)] (conj for inv)
__device__ __forceinline__ void fft_stages(float2* buf, const float2* twd,
                                           int tid, bool inverse) {
#pragma unroll 1
    for (int s = 0; s < 12; ++s) {        // 4096 = 2^12, radix-2 DIT
        __syncthreads();
        const int half  = 1 << s;
        const int shift = 12 - s;
#pragma unroll 1
        for (int w = tid; w < 2048; w += 256) {
            const int j  = w & (half - 1);
            const int i0 = ((w >> s) << (s + 1)) + j;
            const int i1 = i0 + half;
            float2 wt = twd[j << shift];
            if (inverse) wt.y = -wt.y;
            const float2 b = cmul(buf[i1], wt);
            const float2 a = buf[i0];
            buf[i0] = make_float2(a.x + b.x, a.y + b.y);
            buf[i1] = make_float2(a.x - b.x, a.y - b.y);
        }
    }
}

__device__ __forceinline__ float hinterp(const float* Hb, int k) {
    // pos = k * (129-1)/(4097-1) = k/32, align_corners
    const int i0 = k >> 5;
    const float fr = (float)(k & 31) * (1.0f / 32.0f);
    const float h0 = Hb[i0];
    const float h1 = Hb[i0 < 128 ? i0 + 1 : 128];
    return h0 + (h1 - h0) * fr;
}

__device__ __forceinline__ float2 epik(const float2* twd, int k) {
    // e^{-i*pi*k/4096} for k in [0,4096]
    return (k < 4096) ? twd[k] : make_float2(-1.0f, 0.0f);
}

__device__ __forceinline__ float2 xval(const float2* buf, const float2* twd, int k) {
    // X[k] of the 8192-pt rfft from the 4096-pt complex FFT Z (real packing)
    const float2 a = buf[k & 4095];
    float2 b = buf[(4096 - k) & 4095];
    b.y = -b.y;                                   // conj(Z[N-k])
    const float2 u = make_float2(0.5f * (a.x + b.x), 0.5f * (a.y + b.y));
    const float2 v = make_float2(0.5f * (a.x - b.x), 0.5f * (a.y - b.y));
    const float2 t = cmul(epik(twd, k), v);
    return make_float2(u.x + t.y, u.y - t.x);     // u - i*t
}

__global__ __launch_bounds__(256) void fft_filter_kernel(const float* __restrict__ x,
                                                         const float* __restrict__ means,
                                                         const float* __restrict__ Hw,
                                                         float* __restrict__ y) {
    __shared__ float2 buf[4096];                  // 32 KB
    __shared__ float2 twd[4096];                  // 32 KB twiddle table
    const int sig = blockIdx.x;
    const int tid = threadIdx.x;
    const float* xp = x + (size_t)sig * LSIG;
    float*       yp = y + (size_t)sig * LSIG;
    const float* Hb = Hw + (size_t)sig * FR;
    const float  mu = means[sig];

    // twiddle table: e^{-i*pi*m/4096}
    for (int m = tid; m < 4096; m += 256) {
        float sn, cs;
        sincosf(-PI_F * (float)m * (1.0f / 4096.0f), &sn, &cs);
        twd[m] = make_float2(cs, sn);
    }

    // pack z[t] = xc[2t] + i*xc[2t+1], bit-reversed scatter for DIT
    for (int i = tid; i < 4096; i += 256) {
        const float2 z = make_float2(xp[2 * i] - mu, xp[2 * i + 1] - mu);
        buf[__brev((unsigned)i) >> 20] = z;
    }
    fft_stages(buf, twd, tid, false);             // forward FFT (natural order out)
    __syncthreads();

    // Hermitian unpack -> apply interpolated H -> Hermitian repack (to regs)
    float2 zr[16];
#pragma unroll 1
    for (int i = 0; i < 16; ++i) {
        const int k = tid + 256 * i;
        const float2 Xk = xval(buf, twd, k);
        const float2 Xm = xval(buf, twd, 4096 - k);
        const float hk = hinterp(Hb, k);
        const float hm = hinterp(Hb, 4096 - k);
        const float2 Yk = make_float2(hk * Xk.x, hk * Xk.y);
        float2 Ym = make_float2(hm * Xm.x, hm * Xm.y);
        Ym.y = -Ym.y;                             // conj(Y[N-k])
        const float2 u = make_float2(0.5f * (Yk.x + Ym.x), 0.5f * (Yk.y + Ym.y));
        const float2 v = make_float2(0.5f * (Yk.x - Ym.x), 0.5f * (Yk.y - Ym.y));
        float2 e = epik(twd, k);                  // e^{+i*pi*k/4096} = conj
        e.y = -e.y;
        const float2 t = cmul(e, v);
        zr[i] = make_float2(u.x - t.y, u.y + t.x); // u + i*t
    }
    __syncthreads();
#pragma unroll 1
    for (int i = 0; i < 16; ++i) {
        const int k = tid + 256 * i;
        buf[__brev((unsigned)k) >> 20] = zr[i];
    }
    fft_stages(buf, twd, tid, true);              // inverse FFT
    __syncthreads();

    const float sc = 1.0f / 4096.0f;
    for (int i = tid; i < 4096; i += 256) {
        const float2 z = buf[i];
        yp[2 * i]     = z.x * sc;
        yp[2 * i + 1] = z.y * sc;
    }
}

// ------------------------------- launcher ----------------------------------
extern "C" void kernel_launch(void* const* d_in, const int* in_sizes, int n_in,
                              void* d_out, int out_size, void* d_ws, size_t ws_size,
                              hipStream_t stream) {
    (void)in_sizes; (void)n_in; (void)out_size; (void)ws_size;
    const float* x = (const float*)d_in[0];
    float* y = (float*)d_out;

    // workspace carve-out (~2.4 MB, float-aligned; B fragments 32-B aligned)
    float* ws      = (float*)d_ws;
    float* means   = ws;                        // 2048
    float* P_hat   = means + N_SIG;             // 2048*129 = 264192
    float* P_ref   = P_hat + (size_t)N_SIG * FR;        // 64*129 = 8256
    float* H_w     = P_ref + 64 * FR;           // 264192
    _Float16* Bhi  = (_Float16*)(H_w + (size_t)N_SIG * FR);
    _Float16* Blo  = Bhi + (size_t)NTILES * 8 * 32 * 16;   // 69632 halves each

    mean_kernel<<<N_SIG, 256, 0, stream>>>(x, means);

    const int binit = (NTILES * 8 * 32 * 16 + 255) / 256;
    dft_init_kernel<<<binit, 256, 0, stream>>>(Bhi, Blo);

    welch_wmma_kernel<<<N_SIG, 32, 0, stream>>>(x, means, Bhi, Blo, P_hat);

    pref_kernel<<<(64 * FR + 255) / 256, 256, 0, stream>>>(P_hat, P_ref);
    hwelch_kernel<<<(N_SIG * FR + 255) / 256, 256, 0, stream>>>(P_hat, P_ref, H_w);

    fft_filter_kernel<<<N_SIG, 256, 0, stream>>>(x, means, H_w, y);
}
